// GQA_43645457662455
// MI455X (gfx1250) — matrix operand order repaired and checked by reading, not compiled, not run on hardware
//
#include <hip/hip_runtime.h>
#include <hip/hip_bf16.h>

typedef _Float16 f16;
typedef __attribute__((ext_vector_type(16))) _Float16 v16h;
typedef __attribute__((ext_vector_type(8)))  _Float16 v8h;
typedef __attribute__((ext_vector_type(8)))  float    v8f;

union V16 { v16h v; v8h h[2]; };
union H2  { f16 h[2]; unsigned u; };

#define DIMV   448
#define HEADS  8
#define KVH    2
#define HDIM   56
#define HDP    64     // padded head dim
#define GQ     4      // query heads per kv head
#define BATCH  2
#define SEQ    4096

// softmax scale folded with log2(e):  (1/sqrt(56)) * log2(e)
#define QSCALE 0.19278835f

// ---------------------------------------------------------------------------
// WMMA fragment loaders (CDNA5 16x16x32 f16 layouts, per ISA 7.12.2)
// ---------------------------------------------------------------------------
__device__ __forceinline__ v16h load_a16(const f16* p0, int row0, int ld, int col0) {
  const int l = threadIdx.x & 31, r = l & 15, hf = l >> 4;
  const f16* p = p0 + (size_t)(row0 + r) * ld + col0 + hf * 8;
  V16 u;
  u.h[0] = *(const v8h*)(p);
  u.h[1] = *(const v8h*)(p + 16);
  return u.v;
}

__device__ __forceinline__ v16h load_b16(const f16* p0, int row0, int ld, int col0) {
  const int l = threadIdx.x & 31, r = l & 15, hf = l >> 4;
  const f16* p = p0 + (size_t)(row0 + r) * ld + col0 + hf * 8;
  V16 u;
  u.h[0] = *(const v8h*)(p);
  u.h[1] = *(const v8h*)(p + 16);
  return u.v;
}

__device__ __forceinline__ v8f wmma_f16(v16h a, v16h b, v8f c) {
  return __builtin_amdgcn_wmma_f32_16x16x32_f16(
      false, a, false, b, (short)0, c, false, false);
}

// max-reduce across the 16 lanes holding one C-tile row
__device__ __forceinline__ float xor_max16(float v) {
  v = fmaxf(v, __int_as_float(__builtin_amdgcn_ds_swizzle(__float_as_int(v), 0x041F)));
  v = fmaxf(v, __int_as_float(__builtin_amdgcn_ds_swizzle(__float_as_int(v), 0x081F)));
  v = fmaxf(v, __int_as_float(__builtin_amdgcn_ds_swizzle(__float_as_int(v), 0x101F)));
  v = fmaxf(v, __int_as_float(__builtin_amdgcn_ds_swizzle(__float_as_int(v), 0x201F)));
  return v;
}

// CDNA5 async global->LDS copy (16B per enabled lane, tracked by ASYNCcnt).
// Low 32 bits of a generic __shared__ pointer are the wave-relative LDS offset.
__device__ __forceinline__ void async_copy16(unsigned lds_addr, const void* gaddr) {
  asm volatile("global_load_async_to_lds_b128 %0, %1, off"
               :: "v"(lds_addr), "v"(gaddr) : "memory");
}

__device__ __forceinline__ void wait_async4() {
#if __has_builtin(__builtin_amdgcn_s_wait_asynccnt)
  __builtin_amdgcn_s_wait_asynccnt(4);
#else
  asm volatile("s_wait_asynccnt 0x4" ::: "memory");
#endif
}

// ---------------------------------------------------------------------------
// f32 -> f16 cast
// ---------------------------------------------------------------------------
__global__ void cast_f32_to_f16_kernel(const float* __restrict__ src,
                                       f16* __restrict__ dst, int n) {
  const int i = blockIdx.x * blockDim.x + threadIdx.x;
  if (i < n) dst[i] = (f16)src[i];
}

// fill hd row 63 of V^T with ones: softmax denominator column
__global__ void fill_ones_kernel(f16* __restrict__ Vt) {
  const int i = blockIdx.x * blockDim.x + threadIdx.x;  // over BATCH*KVH*SEQ
  if (i < BATCH * KVH * SEQ) {
    const int bk = i / SEQ, t = i % SEQ;
    Vt[((size_t)bk * HDP + (HDP - 1)) * SEQ + t] = (f16)1.0f;
  }
}

// slot permutation within a 32-token group (packs P stores in the flash kernel)
__device__ __forceinline__ int slot_of(int t32) {
  return (t32 < 16) ? (2 * t32) : (2 * (t32 - 16) + 1);
}

// ---------------------------------------------------------------------------
// GEMM: Y[m,n] = oscale * sum_k A[m,k] * W[n,k]   (A: MxK f16, W: NxK f16)
// Each wave computes a 32x64 tile: two A fragments share every B fragment
// (8 WMMAs per 6 fragment loads per k-step).
// MODE 0: scatter f16 into padded head-major layout [B, nHeads, T, HDP]  (Q/K)
// MODE 1: dense f32 output [M, N]                                        (proj)
// MODE 2: scatter f16 transposed+permuted  [B, KVH, HDP, T]              (V^T)
// ---------------------------------------------------------------------------
template <int MODE>
__global__ __launch_bounds__(128) void gemm_wmma_kernel(
    const f16* __restrict__ A, const f16* __restrict__ W,
    f16* __restrict__ outH, float* __restrict__ outF,
    int M, int N, int K, int nHeads, float oscale) {
  const int ntiles = (N + 63) >> 6;
  const int wave = blockIdx.x * (blockDim.x >> 5) + (threadIdx.x >> 5);
  if (wave >= (M >> 5) * ntiles) return;
  const int mt = wave / ntiles;
  const int n0 = (wave % ntiles) << 6;
  const int m0 = mt << 5;

  v8f acc[2][4];
#pragma unroll
  for (int mi = 0; mi < 2; ++mi)
#pragma unroll
    for (int t = 0; t < 4; ++t) { v8f z = {0,0,0,0,0,0,0,0}; acc[mi][t] = z; }

  for (int k0 = 0; k0 < K; k0 += 32) {
    const v16h a0 = load_a16(A, m0,      K, k0);
    const v16h a1 = load_a16(A, m0 + 16, K, k0);
#pragma unroll
    for (int t = 0; t < 4; ++t) {
      const int nn = n0 + t * 16;
      if (nn < N) {
        const v16h bf = load_b16(W, nn, K, k0);
        acc[0][t] = wmma_f16(a0, bf, acc[0][t]);
        acc[1][t] = wmma_f16(a1, bf, acc[1][t]);
      }
    }
  }

  const int l = threadIdx.x & 31, r = l & 15, hf = l >> 4;
#pragma unroll
  for (int mi = 0; mi < 2; ++mi) {
#pragma unroll
    for (int t = 0; t < 4; ++t) {
      const int nn = n0 + t * 16;
      if (nn >= N) continue;
      const int n = nn + r;
#pragma unroll
      for (int i = 0; i < 8; ++i) {
        const int m = m0 + mi * 16 + i + 8 * hf;
        const float v = acc[mi][t][i] * oscale;
        if (MODE == 1) {
          outF[(size_t)m * N + n] = v;
        } else if (MODE == 0) {
          const int bb = m / SEQ, tt = m % SEQ;
          const int hh = n / HDIM, hd = n % HDIM;
          outH[(((size_t)bb * nHeads + hh) * SEQ + tt) * (size_t)HDP + hd] = (f16)v;
        } else {  // MODE 2: V^T, slot-permuted tokens
          const int bb = m / SEQ, tt = m % SEQ;
          const int kvh = n / HDIM, hd = n % HDIM;
          const int pt = (tt & ~31) | slot_of(tt & 31);
          outH[(((size_t)bb * KVH + kvh) * HDP + hd) * (size_t)SEQ + pt] = (f16)v;
        }
      }
    }
  }
}

// ---------------------------------------------------------------------------
// Fused causal flash attention with async double-buffered K/V staging.
// Grid: (SEQ/128, HEADS, BATCH). Block: 128 threads = 4 waves.
// Wave w owns 32 query rows (2 fragment groups of 16).
// K tile: raw contiguous async copy [32 keys x 64 hd].
// V tile: raw async copy of pre-transposed, pre-permuted V^T rows; row hd=63
//         holds ones -> softmax denominator rides in o[3].
// ---------------------------------------------------------------------------
__global__ __launch_bounds__(128) void flash_attn_kernel(
    const f16* __restrict__ Qp, const f16* __restrict__ Kp,
    const f16* __restrict__ Vt, f16* __restrict__ Y) {
  __shared__ f16 Klds[2][32][HDP];       // [buf][key][hd]
  __shared__ f16 Vtlds[2][HDP][32];      // [buf][hd][slot]
  __shared__ unsigned Plds32[4][16][16]; // per-wave packed P

  const int bx = blockIdx.x;
  const int h  = blockIdx.y;
  const int b  = blockIdx.z;
  const int kv = h / GQ;
  const int tid = threadIdx.x;
  const int w  = tid >> 5;
  const int l  = tid & 31, r = l & 15, hf = l >> 4;

  const f16* Kgl  = Kp + (size_t)(b * KVH + kv) * SEQ * HDP;
  const f16* Vtgl = Vt + (size_t)(b * KVH + kv) * HDP * SEQ;
  const size_t qbase = (size_t)(b * HEADS + h) * SEQ * HDP;

  const unsigned kldsA[2] = { (unsigned)(size_t)&Klds[0][0][0],
                              (unsigned)(size_t)&Klds[1][0][0] };
  const unsigned vldsA[2] = { (unsigned)(size_t)&Vtlds[0][0][0],
                              (unsigned)(size_t)&Vtlds[1][0][0] };

  const int q0w = bx * 128 + w * 32;

  v16h qa[2][2];
#pragma unroll
  for (int g = 0; g < 2; ++g) {
    qa[g][0] = load_a16(Qp + qbase, q0w + g * 16, HDP, 0);
    qa[g][1] = load_a16(Qp + qbase, q0w + g * 16, HDP, 32);
  }

  v8f o[2][4];
  float mrow[2][8];
#pragma unroll
  for (int g = 0; g < 2; ++g) {
#pragma unroll
    for (int t = 0; t < 4; ++t) { v8f z = {0,0,0,0,0,0,0,0}; o[g][t] = z; }
#pragma unroll
    for (int i = 0; i < 8; ++i) mrow[g][i] = -1e30f;
  }

  // async staging of one 32-key tile: 4 instructions per wave (16B per lane)
  auto stage = [&](int kb, int buf) {
    const char* kgp = (const char*)(Kgl + (size_t)kb * HDP);   // 4KB contiguous
    const char* vgp = (const char*)(Vtgl + kb);                // 64 rows x 64B
#pragma unroll
    for (int i = 0; i < 2; ++i) {
      const int c = tid + 128 * i;                // chunk id, 16B chunks
      async_copy16(kldsA[buf] + c * 16, kgp + c * 16);
    }
#pragma unroll
    for (int i = 0; i < 2; ++i) {
      const int c = tid + 128 * i;
      const int row = c >> 2, seg = c & 3;
      async_copy16(vldsA[buf] + row * 64 + seg * 16,
                   vgp + (size_t)row * (SEQ * sizeof(f16)) + seg * 16);
    }
  };

  const int kend = bx * 128 + 128;   // causal bound for this block
  stage(0, 0);
  int buf = 0;

  for (int kb = 0; kb < kend; kb += 32) {
    int nkb = kb + 32;
    if (nkb >= kend) nkb = kend - 32;       // clamped redundant prefetch at tail
    stage(nkb, buf ^ 1);
    wait_async4();                          // current tile's 4 async ops done
    __syncthreads();

#pragma unroll
    for (int g = 0; g < 2; ++g) {
      const int q0 = q0w + g * 16;
      if (kb <= q0 + 15) {                  // wave-uniform: skip masked tiles
        const f16* kt = &Klds[buf][0][0];
        v8f s[2];
#pragma unroll
        for (int e = 0; e < 2; ++e) {
          v8f c = {0,0,0,0,0,0,0,0};
          c = wmma_f16(qa[g][0], load_b16(kt, e * 16, HDP, 0),  c);
          c = wmma_f16(qa[g][1], load_b16(kt, e * 16, HDP, 32), c);
          s[e] = c;
        }
        if (kb + 31 > q0) {                 // diagonal tile only
#pragma unroll
          for (int e = 0; e < 2; ++e) {
            const int kk = kb + e * 16 + r;
#pragma unroll
            for (int i = 0; i < 8; ++i) {
              const int qq = q0 + i + 8 * hf;
              s[e][i] = (kk <= qq) ? s[e][i] : -1e30f;
            }
          }
        }
#pragma unroll
        for (int i = 0; i < 8; ++i) {
          const float rm   = xor_max16(fmaxf(s[0][i], s[1][i]));
          const float mnew = fmaxf(mrow[g][i], rm);
          const float alpha = __builtin_amdgcn_exp2f(mrow[g][i] - mnew);
          mrow[g][i] = mnew;
          s[0][i] = __builtin_amdgcn_exp2f(s[0][i] - mnew);
          s[1][i] = __builtin_amdgcn_exp2f(s[1][i] - mnew);
#pragma unroll
          for (int t = 0; t < 4; ++t) o[g][t][i] *= alpha;
        }
#pragma unroll
        for (int i = 0; i < 8; ++i) {
          H2 pk;
          pk.h[0] = (f16)s[0][i];           // key r    -> slot 2r
          pk.h[1] = (f16)s[1][i];           // key 16+r -> slot 2r+1
          Plds32[w][i + 8 * hf][r] = pk.u;
        }
        asm volatile("s_wait_dscnt 0" ::: "memory");
        const v16h pa = load_a16((const f16*)&Plds32[w][0][0], 0, 32, 0);
        const f16* vt = &Vtlds[buf][0][0];
#pragma unroll
        for (int t = 0; t < 4; ++t)
          o[g][t] = wmma_f16(pa, load_b16(vt, t * 16, 32, 0), o[g][t]);
      }
    }
    __syncthreads();                        // all waves done reading buf
    buf ^= 1;
  }

  // epilogue: l lives in o[g][3] lane 15/31; broadcast, normalize, store
#pragma unroll
  for (int g = 0; g < 2; ++g) {
    const int q0 = q0w + g * 16;
#pragma unroll
    for (int i = 0; i < 8; ++i) {
      const float li  = __shfl(o[g][3][i], (l & 16) + 15, 32);
      const float inv = __builtin_amdgcn_rcpf(li);
      const int qq = q0 + i + 8 * hf;
      f16* yrow = Y + (size_t)(b * SEQ + qq) * DIMV + h * HDIM;
#pragma unroll
      for (int t = 0; t < 4; ++t) {
        const int col = t * 16 + r;
        if (col < HDIM) yrow[col] = (f16)(o[g][t][i] * inv);
      }
    }
  }
}

// ---------------------------------------------------------------------------
// Host side
// ---------------------------------------------------------------------------
extern "C" void kernel_launch(void* const* d_in, const int* in_sizes, int n_in,
                              void* d_out, int out_size, void* d_ws, size_t ws_size,
                              hipStream_t stream) {
  const float* x  = (const float*)d_in[0];
  const float* Wq = (const float*)d_in[1];
  const float* Wk = (const float*)d_in[2];
  const float* Wv = (const float*)d_in[3];
  const float* Wp = (const float*)d_in[4];

  const int M = BATCH * SEQ;                       // 8192 tokens
  const size_t nX   = (size_t)M * DIMV;
  const size_t nWq  = (size_t)DIMV * DIMV;
  const size_t nWkv = (size_t)(KVH * HDIM) * DIMV;
  const size_t nQp  = (size_t)BATCH * HEADS * SEQ * HDP;
  const size_t nKp  = (size_t)BATCH * KVH * SEQ * HDP;

  char* p = (char*)d_ws;
  auto take = [&](size_t elems) {
    f16* ret = (f16*)p;
    p += ((elems * sizeof(f16) + 255) & ~(size_t)255);
    return ret;
  };
  f16* xh  = take(nX);
  f16* WqH = take(nWq);
  f16* WkH = take(nWkv);
  f16* WvH = take(nWkv);
  f16* WpH = take(nWq);
  f16* Qp  = take(nQp);
  f16* Kp  = take(nKp);
  f16* Vt  = take(nKp);          // V^T: [B, KVH, HDP, SEQ]
  f16* Yh  = take(nX);

  auto cast = [&](const float* s, f16* d, size_t n) {
    cast_f32_to_f16_kernel<<<(unsigned)((n + 255) / 256), 256, 0, stream>>>(s, d, (int)n);
  };
  cast(x,  xh,  nX);
  cast(Wq, WqH, nWq);
  cast(Wk, WkH, nWkv);
  cast(Wv, WvH, nWkv);
  cast(Wp, WpH, nWq);

  // zero pads (Q/K hd 56..63, V^T rows 56..62), then plant the ones row (63)
  hipMemsetAsync(Qp, 0, nQp * sizeof(f16), stream);
  hipMemsetAsync(Kp, 0, nKp * sizeof(f16), stream);
  hipMemsetAsync(Vt, 0, nKp * sizeof(f16), stream);
  fill_ones_kernel<<<(BATCH * KVH * SEQ + 255) / 256, 256, 0, stream>>>(Vt);

  auto blocks_for = [](int Mm, int Nn) {
    int waves = (Mm >> 5) * ((Nn + 63) >> 6);
    return (unsigned)((waves + 3) / 4);
  };

  gemm_wmma_kernel<0><<<blocks_for(M, DIMV), 128, 0, stream>>>(
      xh, WqH, Qp, nullptr, M, DIMV, DIMV, HEADS, QSCALE);
  gemm_wmma_kernel<0><<<blocks_for(M, KVH * HDIM), 128, 0, stream>>>(
      xh, WkH, Kp, nullptr, M, KVH * HDIM, DIMV, KVH, 1.0f);
  gemm_wmma_kernel<2><<<blocks_for(M, KVH * HDIM), 128, 0, stream>>>(
      xh, WvH, Vt, nullptr, M, KVH * HDIM, DIMV, KVH, 1.0f);

  flash_attn_kernel<<<dim3(SEQ / 128, HEADS, BATCH), 128, 0, stream>>>(Qp, Kp, Vt, Yh);

  gemm_wmma_kernel<1><<<blocks_for(M, DIMV), 128, 0, stream>>>(
      Yh, WpH, nullptr, (float*)d_out, M, DIMV, DIMV, 0, 1.0f);
}
